// AdaptivePIDNetworkControllerV2_71330816851989
// MI455X (gfx1250) — compile-verified, matrix operand order checked by reading, not run on hardware
//
#include <hip/hip_runtime.h>

typedef __attribute__((ext_vector_type(16))) _Float16 v16h;
typedef __attribute__((ext_vector_type(8)))  float    v8f;
typedef __attribute__((ext_vector_type(4)))  float    v4f;

#define PID_STEPS 25
#define PID_DT    0.1f
#define PID_ALPHA 0.5f

// Wave32 CDNA5 layout (CDNA5 ISA):
//  A (16x32 f16, MxK): lane l<16 -> row M=l, elems 0..7 = K 0..7,  elems 8..15 = K 16..23
//                      lane l>=16 -> row M=l-16, elems 0..7 = K 8..15, elems 8..15 = K 24..31
//  B (32x16 f16, KxN): symmetric, column N = lane&15, same K slicing.
//  C/D (16x16 f32):    VGPR v, lane l -> row M = v + 8*(l>=16), col N = l&15.

__launch_bounds__(256, 2)
__global__ void pid_mlp_kernel(
    const float* __restrict__ x,
    const float* __restrict__ W1, const float* __restrict__ b1,
    const float* __restrict__ W2, const float* __restrict__ b2,
    const float* __restrict__ W3, const float* __restrict__ b3,
    const float* __restrict__ Wd1, const float* __restrict__ bd1,
    const float* __restrict__ Wd2, const float* __restrict__ bd2,
    const float* __restrict__ Kp,  const float* __restrict__ Ki,
    const float* __restrict__ Kd,
    float* __restrict__ out, int nSamples)
{
    __shared__ float sc[8][16][17];   // per-wave 16x16 transpose scratch (pad 17 vs bank conflicts)

    const int tid  = threadIdx.x;
    const int lane = tid & 31;
    const int wave = tid >> 5;
    const int n    = lane & 15;       // B/C column index
    const int hi   = lane >> 4;       // half-wave select
    const int ko   = hi << 3;         // K-slice base (0 or 8)
    const int sbase = (blockIdx.x * 8 + wave) * 16;

    // ---------------- Layer 1: h1 = relu(x @ W1^T + b1)  (16x32 * 32x16) ----------------
    v16h a1, bw1;
    {
        const float* xr  = x  + (size_t)(sbase + n) * 32 + ko;  // A row m == lane&15
        const float* w1r = W1 + (size_t)n * 32 + ko;            // B col n == W1 row n
        #pragma unroll
        for (int i = 0; i < 8; ++i) {
            a1[i]      = (_Float16)xr[i];
            a1[i + 8]  = (_Float16)xr[i + 16];
            bw1[i]     = (_Float16)w1r[i];
            bw1[i + 8] = (_Float16)w1r[i + 16];
        }
    }
    v8f c1 = {};
    c1 = __builtin_amdgcn_wmma_f32_16x16x32_f16(false, a1, false, bw1,
                                                (short)0, c1, false, false);
    {
        const float bias = b1[n];
        #pragma unroll
        for (int v = 0; v < 8; ++v) c1[v] = fmaxf(c1[v] + bias, 0.0f);
    }
    #pragma unroll
    for (int v = 0; v < 8; ++v) sc[wave][v + 8 * hi][n] = c1[v];   // C-layout -> LDS
    __syncthreads();

    // ---------------- Layer 2: h2 = relu(h1 @ W2^T + b2)  (K=16 padded to 32, N=8 padded) ----
    // Unconditional loads from a clamped (always-valid) row + 0/1 mask: no EXEC branching.
    v16h a2, bw2;
    {
        const float  nmask = (n < 8) ? 1.0f : 0.0f;
        const float* w2r   = W2 + (size_t)(n & 7) * 16 + ko;      // clamped, always in-bounds
        #pragma unroll
        for (int i = 0; i < 8; ++i) {
            a2[i]      = (_Float16)sc[wave][n][ko + i];           // row m = lane&15
            a2[i + 8]  = (_Float16)0.0f;                          // K 16..31 pad
            bw2[i]     = (_Float16)(w2r[i] * nmask);              // N>=8 pad via mask
            bw2[i + 8] = (_Float16)0.0f;
        }
    }
    v8f c2 = {};
    c2 = __builtin_amdgcn_wmma_f32_16x16x32_f16(false, a2, false, bw2,
                                                (short)0, c2, false, false);
    {
        const float bias = (n < 8) ? 1.0f : 0.0f;  // mask
        const float b2v  = b2[n & 7] * bias;
        #pragma unroll
        for (int v = 0; v < 8; ++v) c2[v] = fmaxf(c2[v] + b2v, 0.0f);
    }
    __syncthreads();
    #pragma unroll
    for (int v = 0; v < 8; ++v) sc[wave][v + 8 * hi][n] = c2[v];
    __syncthreads();

    // ---------------- Layer 3: static_sp = h2 @ W3^T + b3  (K=8 padded, N=8 padded) --------
    v16h a3, bw3;
    {
        const float  amask = (hi == 0) ? 1.0f : 0.0f;             // K-slice 8..15 is all pad
        const float  bmask = ((n < 8) && (hi == 0)) ? 1.0f : 0.0f;
        const float* w3r   = W3 + (size_t)(n & 7) * 8;            // clamped, always in-bounds
        #pragma unroll
        for (int i = 0; i < 8; ++i) {
            a3[i]      = (_Float16)(sc[wave][n][i] * amask);
            a3[i + 8]  = (_Float16)0.0f;
            bw3[i]     = (_Float16)(w3r[i] * bmask);
            bw3[i + 8] = (_Float16)0.0f;
        }
    }
    v8f c3 = {};
    c3 = __builtin_amdgcn_wmma_f32_16x16x32_f16(false, a3, false, bw3,
                                                (short)0, c3, false, false);
    {
        const float b3v = b3[n & 7] * ((n < 8) ? 1.0f : 0.0f);
        #pragma unroll
        for (int v = 0; v < 8; ++v) c3[v] += b3v;
    }
    __syncthreads();
    #pragma unroll
    for (int v = 0; v < 8; ++v) sc[wave][v + 8 * hi][n] = c3[v];   // static_sp tile
    __syncthreads();

    // ---------------- PID recurrence: 32 lanes x 4 elems = 16 samples x 8 channels ----------
    const int half = lane & 1;        // which 4 channels
    const int mrow = lane >> 1;       // sample row within tile
    const int nb   = half * 4;        // channel base

    v4f ss, kp, ki, kd;
    #pragma unroll
    for (int e = 0; e < 4; ++e) {
        ss[e] = sc[wave][mrow][nb + e];
        kp[e] = Kp[nb + e];
        ki[e] = Ki[nb + e];
        kd[e] = Kd[nb + e];
    }
    const float ext = x[(size_t)(sbase + mrow) * 32 + 1];

    // wave-uniform dynamics weights
    const float w00 = Wd1[0], w01 = Wd1[1];
    const float w10 = Wd1[2], w11 = Wd1[3];
    const float w20 = Wd1[4], w21 = Wd1[5];
    const float c0  = bd1[0], c1b = bd1[1], c2b = bd1[2];
    const float z0  = Wd2[0], z1  = Wd2[1], z2  = Wd2[2];
    const float zb  = bd2[0];
    const float extc = PID_ALPHA * ext;

    v4f meas  = {0.f, 0.f, 0.f, 0.f};
    v4f integ = {0.f, 0.f, 0.f, 0.f};
    v4f prev  = {0.f, 0.f, 0.f, 0.f};
    v4f dyn   = ss;

    for (int t = 0; t < PID_STEPS; ++t) {
        v4f err = dyn - meas;
        integ = integ + err * PID_DT;
        #pragma unroll
        for (int e = 0; e < 4; ++e) integ[e] = fminf(fmaxf(integ[e], -5.0f), 5.0f);
        v4f deriv = (err - prev) * (1.0f / PID_DT);
        meas = kp * err + ki * integ + kd * deriv;

        v4f h0  = ss * w00 + meas * w01 + c0;
        v4f h1v = ss * w10 + meas * w11 + c1b;
        v4f h2v = ss * w20 + meas * w21 + c2b;
        #pragma unroll
        for (int e = 0; e < 4; ++e) {
            h0[e]  = fmaxf(h0[e],  0.0f);
            h1v[e] = fmaxf(h1v[e], 0.0f);
            h2v[e] = fmaxf(h2v[e], 0.0f);
        }
        v4f du = h0 * z0 + h1v * z1 + h2v * z2 + zb;
        dyn = du * (1.0f - PID_ALPHA) + extc;
        prev = err;
    }

    // mean over 8 channels: pair (2k, 2k+1) covers row k
    float s = meas[0] + meas[1] + meas[2] + meas[3];
    s += __shfl_xor(s, 1, 32);
    if (half == 0) {
        const int row = sbase + mrow;
        if (row < nSamples) out[row] = s * 0.125f;
    }
}

extern "C" void kernel_launch(void* const* d_in, const int* in_sizes, int n_in,
                              void* d_out, int out_size, void* d_ws, size_t ws_size,
                              hipStream_t stream) {
    const float* x   = (const float*)d_in[0];
    const float* W1  = (const float*)d_in[1];
    const float* b1  = (const float*)d_in[2];
    const float* W2  = (const float*)d_in[3];
    const float* b2  = (const float*)d_in[4];
    const float* W3  = (const float*)d_in[5];
    const float* b3  = (const float*)d_in[6];
    const float* Wd1 = (const float*)d_in[7];
    const float* bd1 = (const float*)d_in[8];
    const float* Wd2 = (const float*)d_in[9];
    const float* bd2 = (const float*)d_in[10];
    const float* Kp  = (const float*)d_in[11];
    const float* Ki  = (const float*)d_in[12];
    const float* Kd  = (const float*)d_in[13];
    float* out = (float*)d_out;

    const int nSamples = in_sizes[0] / 32;      // B = 1048576
    const int blocks   = nSamples / 128;        // 8 waves x 16 samples per block

    pid_mlp_kernel<<<blocks, 256, 0, stream>>>(x, W1, b1, W2, b2, W3, b3,
                                               Wd1, bd1, Wd2, bd2, Kp, Ki, Kd,
                                               out, nSamples);
}
